// MemoryModel_35270271435205
// MI455X (gfx1250) — matrix operand order repaired and checked by reading, not compiled, not run on hardware
//
#include <hip/hip_runtime.h>

typedef __attribute__((ext_vector_type(16))) __bf16 v16bf;
typedef __attribute__((ext_vector_type(8)))  float  v8f;

#define H       32
#define VOCAB   64
#define LSEQ    1024
#define BATCH   2048
#define CHUNK   16
#define WPB     8   // waves per block

__global__ __launch_bounds__(256) void fused_delta_memory_kernel(
    const int*   __restrict__ seq,    // (B, L)
    const float* __restrict__ embed,  // (64, 32)
    const float* __restrict__ w1,     // (64, 32)
    const float* __restrict__ b1,     // (64,)
    const float* __restrict__ w2,     // (32, 64)
    const float* __restrict__ b2,     // (32,)
    const float* __restrict__ ln_g,   // (32,)
    const float* __restrict__ ln_b,   // (32,)
    const float* __restrict__ wr,     // (32, 32)
    const float* __restrict__ br,     // (32,)
    const float* __restrict__ wo,     // (64, 32)
    const float* __restrict__ bo,     // (64,)
    float*       __restrict__ out)    // (B, 64)
{
    // per-wave LDS slices
    __shared__ float  lds_h [WPB][CHUNK * H];    // raw embeddings, token-major
    __shared__ __bf16 lds_u [WPB][CHUNK * 64];   // relu(h@w1.T+b1), bf16
    __shared__ float  lds_ff[WPB][CHUNK * H];    // ff output (pre-residual)

    const int lane  = threadIdx.x & 31;
    const int wave  = threadIdx.x >> 5;
    const int b     = blockIdx.x * WPB + wave;
    if (b >= BATCH) return;                      // wave-uniform

    const int lhalf = lane >> 4;   // 0: lanes 0-15, 1: lanes 16-31
    const int lmod  = lane & 15;

    // ---------------- preload WMMA B operands (resident in VGPRs) ----------
    // B layout (K x 16): lane = column N (mod 16); K = e + 8*(e>=8) + 8*(lane>=16)
    // (same k order as A; 2 bf16 per VGPR).
    v16bf w1B[4];                   // GEMM1: B = w1.T  (32 x 64) -> 4 N-tiles
    #pragma unroll
    for (int nt = 0; nt < 4; ++nt) {
        const int N = nt * 16 + lmod;
        #pragma unroll
        for (int e = 0; e < 16; ++e) {
            const int K = e + (e & 8) + lhalf * 8;
            w1B[nt][e] = (__bf16)w1[N * H + K];
        }
    }
    v16bf w2B[2][2];                // GEMM2: B = w2.T (64 x 32) -> [K-half][N-tile]
    #pragma unroll
    for (int kh = 0; kh < 2; ++kh)
        #pragma unroll
        for (int nt = 0; nt < 2; ++nt) {
            const int N = nt * 16 + lmod;
            #pragma unroll
            for (int e = 0; e < 16; ++e) {
                const int K = kh * 32 + (e + (e & 8)) + lhalf * 8;
                w2B[kh][nt][e] = (__bf16)w2[N * 64 + K];
            }
        }

    float b1v[4], b2v[2];
    #pragma unroll
    for (int nt = 0; nt < 4; ++nt) b1v[nt] = b1[nt * 16 + lmod];
    #pragma unroll
    for (int nt = 0; nt < 2; ++nt) b2v[nt] = b2[nt * 16 + lmod];
    const float lngv = ln_g[lane];
    const float lnbv = ln_b[lane];

    // delta-memory state: lane i holds row i of M (32 f32)
    float Mrow[32];
    #pragma unroll
    for (int j = 0; j < 32; ++j) Mrow[j] = 0.0f;
    float qv = 0.0f;                // query hidden vector (lane = channel)

    const int* seq_b = seq + (size_t)b * LSEQ;

    for (int c = 0; c < LSEQ / CHUNK; ++c) {
        // ---- gather 16 token embeddings: build A (16x32 bf16) + stash f32 ----
        const int tok = seq_b[c * CHUNK + lmod];       // lanes L, L+16 share token
        const float* erow = embed + tok * H;
        v16bf a1;
        #pragma unroll
        for (int e = 0; e < 16; ++e) {
            // 16-bit A layout (16x32): K = e + 8*(e>=8) + 8*(lane>=16)
            const int K = e + (e & 8) + lhalf * 8;
            const float v = erow[K];
            a1[e] = (__bf16)v;
            lds_h[wave][lmod * H + K] = v;             // token-major f32 copy
        }

        // ---- GEMM1: u = h @ w1.T (16x64), +b1, relu -> lds_u (bf16) ----
        #pragma unroll
        for (int nt = 0; nt < 4; ++nt) {
            v8f c1 = {};
            c1 = __builtin_amdgcn_wmma_f32_16x16x32_bf16(
                     false, a1, false, w1B[nt], (short)0, c1, false, false);
            #pragma unroll
            for (int r = 0; r < 8; ++r) {
                const int m = r + 8 * lhalf;           // token row (C layout)
                float u = c1[r] + b1v[nt];
                u = u > 0.0f ? u : 0.0f;
                lds_u[wave][m * 64 + nt * 16 + lmod] = (__bf16)u;
            }
        }

        // ---- rebuild A2 (16x64 bf16) in A layout from lds_u ----
        v16bf a2[2];
        #pragma unroll
        for (int h2 = 0; h2 < 2; ++h2)
            #pragma unroll
            for (int e = 0; e < 16; ++e) {
                const int K = h2 * 32 + (e + (e & 8)) + lhalf * 8;
                a2[h2][e] = lds_u[wave][lmod * 64 + K];
            }

        // ---- GEMM2: ff = relu_u @ w2.T (16x32), +b2 -> lds_ff ----
        #pragma unroll
        for (int nt = 0; nt < 2; ++nt) {
            v8f c2 = {};
            c2 = __builtin_amdgcn_wmma_f32_16x16x32_bf16(
                     false, a2[0], false, w2B[0][nt], (short)0, c2, false, false);
            c2 = __builtin_amdgcn_wmma_f32_16x16x32_bf16(
                     false, a2[1], false, w2B[1][nt], (short)0, c2, false, false);
            #pragma unroll
            for (int r = 0; r < 8; ++r) {
                const int m = r + 8 * lhalf;
                lds_ff[wave][m * H + nt * 16 + lmod] = c2[r] + b2v[nt];
            }
        }

        // ---- sequential delta-memory scan over the 16 tokens ----
        for (int tt = 0; tt < CHUNK; ++tt) {
            const int gt = c * CHUNK + tt;
            // residual + LayerNorm (lane = channel)
            const float xv = lds_h[wave][tt * H + lane] + lds_ff[wave][tt * H + lane];
            float s = xv;
            #pragma unroll
            for (int m = 16; m >= 1; m >>= 1) s += __shfl_xor(s, m, 32);
            const float mu = s * (1.0f / 32.0f);
            const float d  = xv - mu;
            float v2 = d * d;
            #pragma unroll
            for (int m = 16; m >= 1; m >>= 1) v2 += __shfl_xor(v2, m, 32);
            const float var = v2 * (1.0f / 32.0f);
            const float k = d * rsqrtf(var + 1e-5f) * lngv + lnbv;

            if (gt == LSEQ - 1) {
                qv = k;                                 // last token = query
            } else {
                float den = k * k;
                #pragma unroll
                for (int m = 16; m >= 1; m >>= 1) den += __shfl_xor(den, m, 32);
                den += 1e-6f;
                float kj[32];
                float vp = 0.0f;
                #pragma unroll
                for (int j = 0; j < 32; ++j) {          // broadcast k once
                    kj[j] = __shfl(k, j, 32);
                    vp = fmaf(Mrow[j], kj[j], vp);      // v_pred[i]
                }
                const float dv = k - vp / den;
                #pragma unroll
                for (int j = 0; j < 32; ++j)            // rank-1 update
                    Mrow[j] = fmaf(dv, kj[j], Mrow[j]);
            }
        }
    }

    // ---- context = M @ q ----
    float ctx = 0.0f;
    #pragma unroll
    for (int j = 0; j < 32; ++j) {
        const float qj = __shfl(qv, j, 32);
        ctx = fmaf(Mrow[j], qj, ctx);
    }
    // ---- read = context @ wr.T + br ----
    float rd = br[lane];
    #pragma unroll
    for (int j = 0; j < 32; ++j) {
        const float cj = __shfl(ctx, j, 32);
        rd = fmaf(wr[lane * H + j], cj, rd);
    }
    // ---- out = read @ wo.T + bo : lane computes vocab entries lane, lane+32 ----
    float o0 = bo[lane];
    float o1 = bo[lane + 32];
    #pragma unroll
    for (int j = 0; j < 32; ++j) {
        const float rj = __shfl(rd, j, 32);
        o0 = fmaf(wo[lane * H + j],        rj, o0);
        o1 = fmaf(wo[(lane + 32) * H + j], rj, o1);
    }
    out[(size_t)b * 64 + lane]      = o0;
    out[(size_t)b * 64 + 32 + lane] = o1;
}

extern "C" void kernel_launch(void* const* d_in, const int* in_sizes, int n_in,
                              void* d_out, int out_size, void* d_ws, size_t ws_size,
                              hipStream_t stream) {
    const int*   seq   = (const int*)  d_in[0];
    const float* embed = (const float*)d_in[1];
    const float* w1    = (const float*)d_in[2];
    const float* b1    = (const float*)d_in[3];
    const float* w2    = (const float*)d_in[4];
    const float* b2    = (const float*)d_in[5];
    const float* ln_g  = (const float*)d_in[6];
    const float* ln_b  = (const float*)d_in[7];
    const float* wr    = (const float*)d_in[8];
    const float* br    = (const float*)d_in[9];
    const float* wo    = (const float*)d_in[10];
    const float* bo    = (const float*)d_in[11];
    float* out = (float*)d_out;

    dim3 grid(BATCH / WPB);   // 256 blocks
    dim3 block(WPB * 32);     // 8 waves of 32
    fused_delta_memory_kernel<<<grid, block, 0, stream>>>(
        seq, embed, w1, b1, w2, b2, ln_g, ln_b, wr, br, wo, bo, out);
}
